// SelfAttention_37417755083432
// MI455X (gfx1250) — matrix operand order repaired and checked by reading, not compiled, not run on hardware
//
#include <hip/hip_runtime.h>
#include <hip/hip_bf16.h>

// ---------------------------------------------------------------------------
// Self-attention for MI455X (gfx1250, wave32, WMMA).
//   D = 1024, D_K = 1024, B = 4, N = 2048
// Pipeline:
//   1) cast x (fp32) -> bf16
//   2) cast+transpose W (fp32 [K=1024][N=3072]) -> Wt bf16 [3072][1024]
//   3) QKV GEMM via v_wmma_f32_16x16x32_bf16, 32x64 wave tile, double-
//      buffered fragment loads; epilogue adds bias and writes Q,K row-major
//      bf16 and V transposed (Vt[d][n]) bf16
//   4) flash attention: per-block 16 queries, 8 waves split d into 128-wide
//      slices; QK^T partials reduced via ds_add_f32; online softmax; P@V
//      with batched V-fragment loads feeding back-to-back WMMAs.
// ---------------------------------------------------------------------------

typedef __attribute__((ext_vector_type(16))) __bf16 v16bf;
typedef __attribute__((ext_vector_type(8)))  float  v8f;

#define BATCH   4
#define SEQ     2048
#define DIM     1024
#define NOUT    3072
#define MROWS   (BATCH * SEQ)          // 8192

union Frag {
    uint4          q[2];
    unsigned int   u32[8];
    unsigned short s[16];
    v16bf          v;
};

__device__ __forceinline__ v8f wmma_bf16(v16bf a, v16bf b, v8f c) {
    return __builtin_amdgcn_wmma_f32_16x16x32_bf16(
        /*neg_a=*/false, a, /*neg_b=*/false, b,
        /*c_mod=*/(short)0, c, /*reuse_a=*/false, /*reuse_b=*/false);
}

__device__ __forceinline__ unsigned short f2bf(float f) {
    unsigned int u = __float_as_uint(f);
    u += 0x7FFFu + ((u >> 16) & 1u);   // round-to-nearest-even
    return (unsigned short)(u >> 16);
}

__device__ __forceinline__ v8f zero8() {
    return v8f{0.f, 0.f, 0.f, 0.f, 0.f, 0.f, 0.f, 0.f};
}

// ---------------------------------------------------------------- cast x
__global__ __launch_bounds__(256) void cast_x_kernel(
    const float* __restrict__ x, unsigned short* __restrict__ xb, int n) {
    int i = blockIdx.x * 256 + threadIdx.x;
    if (i < n) xb[i] = f2bf(x[i]);
}

// ------------------------------------------------- cast + transpose W
__global__ __launch_bounds__(256) void cast_wt_kernel(
    const float* __restrict__ w, unsigned short* __restrict__ wt) {
    int i = blockIdx.x * 256 + threadIdx.x;
    if (i < DIM * NOUT) {
        int k = i / NOUT;          // coalesced read along n
        int n = i % NOUT;
        wt[(size_t)n * DIM + k] = f2bf(w[i]);
    }
}

// ------------------------------------------------------------ QKV GEMM
// block = 256 thr = 8 waves. Wave: 32 rows x 64 cols (2x4 accum tiles).
// Block tile: 256 x 64. Grid: (8192/256, 3072/64) = (32, 48).
// Fragment loads are double-buffered so WMMAs of step i overlap the
// global_load_b128 bursts of step i+1.
__global__ __launch_bounds__(256) void qkv_gemm_kernel(
    const unsigned short* __restrict__ xb,   // [8192][1024] bf16
    const unsigned short* __restrict__ wtb,  // [3072][1024] bf16 (W^T)
    const float*          __restrict__ bias, // [3072]
    unsigned short* __restrict__ qb,         // [8192][1024]
    unsigned short* __restrict__ kb,         // [8192][1024]
    unsigned short* __restrict__ vtb) {      // [4][1024][2048]
    const int wave  = threadIdx.x >> 5;
    const int lane  = threadIdx.x & 31;
    const int half  = lane >> 4;
    const int l15   = lane & 15;
    const int rbase = blockIdx.x * 256 + wave * 32;
    const int nbase = blockIdx.y * 64;

    v8f acc[2][4];
#pragma unroll
    for (int rt = 0; rt < 2; ++rt)
#pragma unroll
        for (int t = 0; t < 4; ++t) acc[rt][t] = zero8();

    const unsigned short* xrow[2];
    xrow[0] = xb + (size_t)(rbase + l15) * DIM;
    xrow[1] = xb + (size_t)(rbase + 16 + l15) * DIM;
    const unsigned short* wcol[4];
#pragma unroll
    for (int t = 0; t < 4; ++t)
        wcol[t] = wtb + (size_t)(nbase + t * 16 + l15) * DIM + half * 8;

    Frag abuf[2][2], bbuf[2][4];

    auto load_step = [&](int ks, int buf) {
#pragma unroll
        for (int rt = 0; rt < 2; ++rt) {
            const unsigned short* ap = xrow[rt] + ks * 32 + half * 8;
            abuf[buf][rt].q[0] = *(const uint4*)(ap);
            abuf[buf][rt].q[1] = *(const uint4*)(ap + 16);
        }
#pragma unroll
        for (int t = 0; t < 4; ++t) {
            const unsigned short* bp = wcol[t] + ks * 32;
            bbuf[buf][t].q[0] = *(const uint4*)(bp);
            bbuf[buf][t].q[1] = *(const uint4*)(bp + 16);
        }
    };
    auto compute_step = [&](int buf) {
#pragma unroll
        for (int rt = 0; rt < 2; ++rt)
#pragma unroll
            for (int t = 0; t < 4; ++t)
                acc[rt][t] = wmma_bf16(abuf[buf][rt].v, bbuf[buf][t].v, acc[rt][t]);
    };

    load_step(0, 0);
    for (int ks = 0; ks < DIM / 32 - 1; ++ks) {
        load_step(ks + 1, (ks + 1) & 1);   // prefetch next k-step
        compute_step(ks & 1);              // compute current k-step
    }
    compute_step((DIM / 32 - 1) & 1);

    // epilogue: + bias, route Q / K / V^T
#pragma unroll
    for (int rt = 0; rt < 2; ++rt) {
#pragma unroll
        for (int t = 0; t < 4; ++t) {
            const int col = nbase + t * 16 + l15;
            const float bv = bias[col];
#pragma unroll
            for (int r = 0; r < 8; ++r) {
                const int m = rbase + rt * 16 + half * 8 + r;
                const unsigned short h = f2bf(acc[rt][t][r] + bv);
                if (col < 1024) {
                    qb[(size_t)m * DIM + col] = h;
                } else if (col < 2048) {
                    kb[(size_t)m * DIM + (col - 1024)] = h;
                } else {
                    const int b   = m >> 11;
                    const int pos = m & 2047;
                    vtb[(size_t)b * DIM * SEQ + (size_t)(col - 2048) * SEQ + pos] = h;
                }
            }
        }
    }
}

// ------------------------------------------------------- flash attention
// block = 256 thr = 8 waves, 16 queries. Wave w owns d-slice [w*128, w*128+128).
// Key loop: 64 tiles of 32 keys; 2 barriers per tile.
__global__ __launch_bounds__(256) void attn_kernel(
    const unsigned short* __restrict__ qb,   // [8192][1024]
    const unsigned short* __restrict__ kb,   // [8192][1024]
    const unsigned short* __restrict__ vtb,  // [4][1024][2048]
    float* __restrict__ out) {               // [4][2048][1024]
    __shared__ float          S[2][16][32];
    __shared__ unsigned short P[16][32];
    __shared__ float          alpha_l[16];
    __shared__ float          l_l[16];

    const int tid   = threadIdx.x;
    const int wave  = tid >> 5;
    const int lane  = tid & 31;
    const int half  = lane >> 4;
    const int l15   = lane & 15;
    const int bi    = blockIdx.y;
    const int qbase = blockIdx.x * 16;
    const int dbase = wave * 128;
    const float scale = 0.03125f;  // 1/sqrt(1024)

    v8f yacc[8];
#pragma unroll
    for (int t = 0; t < 8; ++t) yacc[t] = zero8();

    float m_run = -1.0e30f;  // wave0 lanes 0..15 only
    float l_run = 0.0f;

    {   // zero S[0]
        float* s0 = &S[0][0][0];
        for (int i = tid; i < 512; i += 256) s0[i] = 0.f;
    }
    __syncthreads();

    const unsigned short* qrow = qb + (size_t)(bi * SEQ + qbase + l15) * DIM + dbase;
    const unsigned short* krow = kb + (size_t)(bi * SEQ + l15) * DIM + dbase + half * 8;
    const unsigned short* vbat = vtb + (size_t)bi * DIM * SEQ;

    for (int kt = 0; kt < SEQ / 32; ++kt) {
        const int buf = kt & 1;

        // ---- partial scores over this wave's 128-wide d slice ----
        v8f sacc[2];
        sacc[0] = zero8();
        sacc[1] = zero8();

        Frag qa[2], kf[2][2];
        auto load_sc = [&](int ks, int b) {
            const unsigned short* ap = qrow + ks * 32 + half * 8;
            qa[b].q[0] = *(const uint4*)(ap);
            qa[b].q[1] = *(const uint4*)(ap + 16);
#pragma unroll
            for (int sub = 0; sub < 2; ++sub) {
                const unsigned short* bp =
                    krow + (size_t)(kt * 32 + sub * 16) * DIM + ks * 32;
                kf[b][sub].q[0] = *(const uint4*)(bp);
                kf[b][sub].q[1] = *(const uint4*)(bp + 16);
            }
        };
        load_sc(0, 0);
#pragma unroll
        for (int ks = 0; ks < 4; ++ks) {
            const int b = ks & 1;
            if (ks < 3) load_sc(ks + 1, b ^ 1);   // prefetch next d-chunk
            sacc[0] = wmma_bf16(qa[b].v, kf[b][0].v, sacc[0]);
            sacc[1] = wmma_bf16(qa[b].v, kf[b][1].v, sacc[1]);
        }
#pragma unroll
        for (int sub = 0; sub < 2; ++sub)
#pragma unroll
            for (int r = 0; r < 8; ++r)
                atomicAdd(&S[buf][half * 8 + r][sub * 16 + l15], sacc[sub][r]);
        __syncthreads();

        // ---- wave0: online softmax; waves 1..7: zero other buffer ----
        if (wave == 0) {
            if (lane < 16) {
                const int row = lane;
                float tmax = -1.0e30f;
                for (int j = 0; j < 32; ++j)
                    tmax = fmaxf(tmax, S[buf][row][j] * scale);
                const float m_new = fmaxf(m_run, tmax);
                const float alpha = __expf(m_run - m_new);
                float psum = 0.f;
                for (int j = 0; j < 32; ++j) {
                    const float p = __expf(S[buf][row][j] * scale - m_new);
                    P[row][j] = f2bf(p);
                    psum += p;
                }
                l_run = l_run * alpha + psum;
                m_run = m_new;
                alpha_l[row] = alpha;
            }
        } else {
            float* sz = &S[buf ^ 1][0][0];
            for (int i = tid - 32; i < 512; i += 224) sz[i] = 0.f;
        }
        __syncthreads();

        // ---- rescale y accumulators, then y += P @ V ----
        float al[8];
#pragma unroll
        for (int r = 0; r < 8; ++r) al[r] = alpha_l[half * 8 + r];
#pragma unroll
        for (int t = 0; t < 8; ++t)
#pragma unroll
            for (int r = 0; r < 8; ++r) yacc[t][r] *= al[r];

        Frag p;  // A fragment: 16 queries x 32 keys (bf16)
        {
            const unsigned int* pr = (const unsigned int*)&P[l15][0];
#pragma unroll
            for (int i = 0; i < 4; ++i) p.u32[i]     = pr[half * 4 + i];
#pragma unroll
            for (int i = 0; i < 4; ++i) p.u32[4 + i] = pr[half * 4 + 8 + i];
        }

        Frag bv[8];  // batched V^T fragment loads -> one wait, 8 WMMAs
#pragma unroll
        for (int t = 0; t < 8; ++t) {
            const unsigned short* bp =
                vbat + (size_t)(dbase + t * 16 + l15) * SEQ + kt * 32 + half * 8;
            bv[t].q[0] = *(const uint4*)(bp);
            bv[t].q[1] = *(const uint4*)(bp + 16);
        }
#pragma unroll
        for (int t = 0; t < 8; ++t)
            yacc[t] = wmma_bf16(p.v, bv[t].v, yacc[t]);
    }

    // ---- finalize: y /= l ----
    if (wave == 0 && lane < 16) l_l[lane] = l_run;
    __syncthreads();

    float inv[8];
#pragma unroll
    for (int r = 0; r < 8; ++r) inv[r] = 1.0f / l_l[half * 8 + r];

#pragma unroll
    for (int t = 0; t < 8; ++t) {
        const int col = dbase + t * 16 + l15;
#pragma unroll
        for (int r = 0; r < 8; ++r) {
            const int row = qbase + half * 8 + r;
            out[(size_t)(bi * SEQ + row) * DIM + col] = yacc[t][r] * inv[r];
        }
    }
}

// ---------------------------------------------------------------------------
extern "C" void kernel_launch(void* const* d_in, const int* in_sizes, int n_in,
                              void* d_out, int out_size, void* d_ws, size_t ws_size,
                              hipStream_t stream) {
    const float* x    = (const float*)d_in[0];  // [4][2048][1024]
    const float* W    = (const float*)d_in[1];  // [1024][3072]
    const float* bias = (const float*)d_in[2];  // [3072]
    float*       out  = (float*)d_out;          // [4][2048][1024]

    char* ws = (char*)d_ws;
    const size_t xN = (size_t)MROWS * DIM;               // 8,388,608
    unsigned short* xb  = (unsigned short*)(ws);                         // 16 MB
    unsigned short* wtb = (unsigned short*)(ws + 2 * xN);                //  6 MB
    unsigned short* qb  = (unsigned short*)(ws + 2 * xN + 2 * (size_t)DIM * NOUT);
    unsigned short* kb  = (unsigned short*)((char*)qb + 2 * xN);
    unsigned short* vtb = (unsigned short*)((char*)kb + 2 * xN);

    cast_x_kernel<<<(int)((xN + 255) / 256), 256, 0, stream>>>(x, xb, (int)xN);
    cast_wt_kernel<<<(DIM * NOUT + 255) / 256, 256, 0, stream>>>(W, wtb);
    qkv_gemm_kernel<<<dim3(MROWS / 256, NOUT / 64), 256, 0, stream>>>(
        xb, wtb, bias, qb, kb, vtb);
    attn_kernel<<<dim3(SEQ / 16, BATCH), 256, 0, stream>>>(qb, kb, vtb, out);
}
